// BigramLanguageModel_11544872091817
// MI455X (gfx1250) — matrix operand order repaired and checked by reading, not compiled, not run on hardware
//
#include <hip/hip_runtime.h>
#include <hip/hip_bf16.h>

// ---------------------------------------------------------------------------
// Types for CDNA5 WMMA (wave32): 16x16x32 bf16 -> f32
// ---------------------------------------------------------------------------
typedef __attribute__((ext_vector_type(16))) __bf16 v16bf;
typedef __attribute__((ext_vector_type(8)))  float  v8f;

union Frag16 { v16bf v; uint4 q[2]; };

__device__ __forceinline__ v8f wmma_bf16(v16bf a, v16bf b, v8f c) {
  // (neg_a, A, neg_b, B, c_mod, C, reuse_a, reuse_b)
  return __builtin_amdgcn_wmma_f32_16x16x32_bf16(false, a, false, b, (short)0, c,
                                                 false, false);
}

// Async global->LDS 16B copy (gfx1250; GVS mode: SGPR base + VGPR i32 offset).
__device__ __forceinline__ void async_copy_b128(unsigned lds_off, unsigned gvoff,
                                                unsigned long long gbase) {
  asm volatile("global_load_async_to_lds_b128 %0, %1, %2"
               :: "v"(lds_off), "v"(gvoff), "s"(gbase) : "memory");
}
__device__ __forceinline__ void wait_asynccnt0() {
#if __has_builtin(__builtin_amdgcn_s_wait_asynccnt)
  __builtin_amdgcn_s_wait_asynccnt(0);
#else
  asm volatile("s_wait_asynccnt 0x0" ::: "memory");
#endif
}

// Model dims
#define BB 2
#define TT 1024
#define DD 1024
#define HH 16
#define HD 64
#define LL 6
#define VV 32000
#define BT (BB * TT)

// ---------------------------------------------------------------------------
// Generic batched GEMM: C[z] = op(A[z] @ B[z] + bias (+ Res)), bf16 WMMA math.
// Block: 128 threads = 4 waves; block tile 64x64; K-tile 64; wave tile 32x32.
// LDS layouts: every WMMA fragment = 2 x 16B contiguous ds_load_b128.
// out_bf16: store C as bf16 (used for q/k/v so attention can async-DMA tiles).
// Requires M%64==0, N%64==0, K%64==0 (true for all launches here).
// ---------------------------------------------------------------------------
__global__ __launch_bounds__(128)
void gemm_bf16_kernel(const float* __restrict__ A, const float* __restrict__ Bm,
                      const float* __restrict__ bias, const float* __restrict__ Res,
                      float* __restrict__ C,
                      int M, int N, int K, int lda, int ldb, int ldc,
                      long sAz, long sBz, long sCz, int relu, int out_bf16)
{
  __shared__ __align__(16) __bf16 As [64][72];   // 64 rows x 64 K (+pad, 144B stride)
  __shared__ __align__(16) __bf16 BsT[64][72];   // 64 N    x 64 K (transposed, +pad)

  const int z = blockIdx.z;
  A  += (long)z * sAz;
  Bm += (long)z * sBz;
  if (Res) Res += (long)z * sCz;

  const int tid  = threadIdx.x;
  const int lane = tid & 31;
  const int wave = tid >> 5;
  const int wr   = wave >> 1;
  const int wc   = wave & 1;
  const int ml   = lane & 15;
  const int hl   = lane >> 4;
  const int m0   = blockIdx.y * 64;
  const int n0   = blockIdx.x * 64;

  v8f acc[2][2] = {};

  for (int k0 = 0; k0 < K; k0 += 64) {
    // ---- Stage A tile (64x64) row-major as bf16, vector LDS stores.
    {
      const int r = tid >> 1, c = (tid & 1) * 32;
      const float* src = A + (long)(m0 + r) * lda + (k0 + c);
      __align__(16) __bf16 ta[32];
#pragma unroll
      for (int i = 0; i < 32; ++i) ta[i] = (__bf16)src[i];
#pragma unroll
      for (int i = 0; i < 4; ++i)
        *(uint4*)&As[r][c + i * 8] = ((const uint4*)ta)[i];

      // ---- Stage B tile transposed: BsT[n][kk] = B[k0+kk][n0+n].
      const int n = tid >> 1, kc = (tid & 1) * 32;
      const float* srcb = Bm + (long)(k0 + kc) * ldb + (n0 + n);
      __align__(16) __bf16 tb[32];
#pragma unroll
      for (int i = 0; i < 32; ++i) tb[i] = (__bf16)srcb[(long)i * ldb];
#pragma unroll
      for (int i = 0; i < 4; ++i)
        *(uint4*)&BsT[n][kc + i * 8] = ((const uint4*)tb)[i];

      // ---- Prefetch next k-tile into cache while WMMAs run.
      if (k0 + 64 < K) {
        __builtin_prefetch(src + 64, 0, 1);                 // global_prefetch_b8
        __builtin_prefetch(srcb + (long)64 * ldb, 0, 1);
      }
    }
    __syncthreads();

#pragma unroll
    for (int ks = 0; ks < 2; ++ks) {            // two 32-K WMMA steps
      const int kb = ks * 32;
      Frag16 a[2], b[2];
#pragma unroll
      for (int i = 0; i < 2; ++i) {
        const int m = wr * 32 + i * 16 + ml;
        // A layout: lo lanes K 0-7 & 16-23, hi lanes K 8-15 & 24-31.
        a[i].q[0] = *(const uint4*)&As[m][kb + hl * 8];
        a[i].q[1] = *(const uint4*)&As[m][kb + 16 + hl * 8];
      }
#pragma unroll
      for (int j = 0; j < 2; ++j) {
        const int n = wc * 32 + j * 16 + ml;
        // B layout: lo lanes K 0-7 & 16-23, hi lanes K 8-15 & 24-31 (matches A).
        b[j].q[0] = *(const uint4*)&BsT[n][kb + hl * 8];
        b[j].q[1] = *(const uint4*)&BsT[n][kb + 16 + hl * 8];
      }
#pragma unroll
      for (int i = 0; i < 2; ++i)
#pragma unroll
        for (int j = 0; j < 2; ++j)
          acc[i][j] = wmma_bf16(a[i].v, b[j].v, acc[i][j]);
    }
    __syncthreads();
  }

  // Epilogue: C layout = N per lane, M = half*8 + vgpr.
  float*  Cf = C + (long)z * sCz;
  __bf16* Ch = (__bf16*)C + (long)z * sCz;
#pragma unroll
  for (int i = 0; i < 2; ++i) {
#pragma unroll
    for (int j = 0; j < 2; ++j) {
      const int n  = n0 + wc * 32 + j * 16 + ml;
      const float bv = bias ? bias[n] : 0.0f;
#pragma unroll
      for (int r = 0; r < 8; ++r) {
        const int m = m0 + wr * 32 + i * 16 + hl * 8 + r;
        float val = acc[i][j][r] + bv;
        if (Res)  val += Res[(long)m * ldc + n];
        if (relu) val = val > 0.0f ? val : 0.0f;
        if (out_bf16) Ch[(long)m * ldc + n] = (__bf16)val;
        else          Cf[(long)m * ldc + n] = val;
      }
    }
  }
}

// ---------------------------------------------------------------------------
// Causal attention (flash style), per (b,h). Block = 128 thr = 4 waves,
// block covers 64 query rows (16 per wave). HD = 64. scale = D^-0.5 = 1/32.
// q/k/v are bf16. K,V tiles staged into LDS with global_load_async_to_lds_b128
// (ASYNCcnt), V transposed LDS->LDS for contiguous P@V B-fragments.
// ---------------------------------------------------------------------------
__global__ __launch_bounds__(128)
void attn_kernel(const __bf16* __restrict__ q, const __bf16* __restrict__ k,
                 const __bf16* __restrict__ v, float* __restrict__ o, float scale)
{
  __shared__ __align__(16) __bf16 kl [64][72];      // [key][e]  (async DMA dst)
  __shared__ __align__(16) __bf16 vl [64][72];      // [key][e]  (async DMA dst)
  __shared__ __align__(16) __bf16 vlT[64][72];      // [e][key]  (LDS transpose)
  __shared__ __align__(16) __bf16 pl [4][16][72];   // per-wave [q-row][key]

  const int zz = blockIdx.y;           // b*H + h
  const int b  = zz >> 4;
  const int h  = zz & 15;
  const long base = ((long)b * TT) * DD + (long)h * HD;
  const __bf16* qp = q + base;
  const __bf16* kp = k + base;
  const __bf16* vp = v + base;
  float*        op = o + base;

  const int tid  = threadIdx.x;
  const int lane = tid & 31;
  const int wave = tid >> 5;
  const int ml   = lane & 15;
  const int hl   = lane >> 4;
  const int qrow0 = blockIdx.x * 64 + wave * 16;

  // Q fragments (A layout) straight from global bf16: 2 x b128 each.
  Frag16 aq[2];
#pragma unroll
  for (int ks = 0; ks < 2; ++ks) {
    const __bf16* qr = qp + (long)(qrow0 + ml) * DD + ks * 32;
    aq[ks].q[0] = *(const uint4*)(qr + hl * 8);
    aq[ks].q[1] = *(const uint4*)(qr + 16 + hl * 8);
  }

  v8f oacc[4] = {};
  float mrow[8], lrow[8];
#pragma unroll
  for (int r = 0; r < 8; ++r) { mrow[r] = -1e30f; lrow[r] = 0.0f; }

  // Per-thread DMA assignment: row = tid>>1, 32 bf16 (64B) per matrix.
  const int arow = tid >> 1, acol = (tid & 1) * 32;
  const unsigned kl_off = (unsigned)(unsigned long long)(const void*)&kl[arow][acol];
  const unsigned vl_off = (unsigned)(unsigned long long)(const void*)&vl[arow][acol];
  const unsigned long long kb64 = (unsigned long long)kp;
  const unsigned long long vb64 = (unsigned long long)vp;

  for (int kt = 0; kt <= blockIdx.x; ++kt) {
    // ---- Async DMA K,V tiles (key-major bf16) into LDS.
    {
      const unsigned gvo = (unsigned)(((long)(kt * 64 + arow) * DD + acol) * 2);
#pragma unroll
      for (int i = 0; i < 4; ++i) {
        async_copy_b128(kl_off + i * 16u, gvo + i * 16u, kb64);
        async_copy_b128(vl_off + i * 16u, gvo + i * 16u, vb64);
      }
      if (kt < blockIdx.x) {   // prefetch next tile into L2
        __builtin_prefetch(kp + (long)(kt * 64 + 64 + arow) * DD + acol, 0, 1);
        __builtin_prefetch(vp + (long)(kt * 64 + 64 + arow) * DD + acol, 0, 1);
      }
    }
    wait_asynccnt0();
    __syncthreads();

    // ---- LDS transpose V: vlT[e][key] = vl[key][e] (vector stores).
    {
      const int e = tid >> 1, keyc = (tid & 1) * 32;
      __align__(16) __bf16 tv[32];
#pragma unroll
      for (int i = 0; i < 32; ++i) tv[i] = vl[keyc + i][e];
#pragma unroll
      for (int i = 0; i < 4; ++i)
        *(uint4*)&vlT[e][keyc + i * 8] = ((const uint4*)tv)[i];
    }
    __syncthreads();

    // ---- Scores: S = Q @ K^T  (16 q-rows x 64 keys per wave).
    float s[4][8];
#pragma unroll
    for (int j = 0; j < 4; ++j) {
      v8f sacc = {};
      const int n = j * 16 + ml;            // key within tile
#pragma unroll
      for (int ks = 0; ks < 2; ++ks) {
        Frag16 bk;                          // B[kk=e, n=key] = K[key, e]
        bk.q[0] = *(const uint4*)&kl[n][ks * 32 + hl * 8];
        bk.q[1] = *(const uint4*)&kl[n][ks * 32 + 16 + hl * 8];
        sacc = wmma_bf16(aq[ks].v, bk.v, sacc);
      }
      const int col = kt * 64 + n;
#pragma unroll
      for (int r = 0; r < 8; ++r) {
        const int row = qrow0 + hl * 8 + r;
        const float sv = sacc[r] * scale;
        s[j][r] = (col <= row) ? sv : -1e30f;     // causal mask
      }
    }

    // ---- Online softmax per row (16 lanes of this half-wave hold one row).
#pragma unroll
    for (int r = 0; r < 8; ++r) {
      float mx = fmaxf(fmaxf(s[0][r], s[1][r]), fmaxf(s[2][r], s[3][r]));
#pragma unroll
      for (int d = 1; d < 16; d <<= 1) mx = fmaxf(mx, __shfl_xor(mx, d, 32));
      const float mnew = fmaxf(mrow[r], mx);
      const float corr = __expf(mrow[r] - mnew);
      float rs = 0.0f;
#pragma unroll
      for (int j = 0; j < 4; ++j) { s[j][r] = __expf(s[j][r] - mnew); rs += s[j][r]; }
#pragma unroll
      for (int d = 1; d < 16; d <<= 1) rs += __shfl_xor(rs, d, 32);
      lrow[r] = lrow[r] * corr + rs;
      mrow[r] = mnew;
#pragma unroll
      for (int j = 0; j < 4; ++j) oacc[j][r] *= corr;
    }

    // ---- Stage P (C layout -> row-major bf16), then O += P @ V.
#pragma unroll
    for (int j = 0; j < 4; ++j)
#pragma unroll
      for (int r = 0; r < 8; ++r)
        pl[wave][hl * 8 + r][j * 16 + ml] = (__bf16)s[j][r];
    // Same-wave DS ops are in-order: P readable below without a barrier.

#pragma unroll
    for (int ks = 0; ks < 2; ++ks) {
      Frag16 ap;                            // A layout over K=keys
      ap.q[0] = *(const uint4*)&pl[wave][ml][ks * 32 + hl * 8];
      ap.q[1] = *(const uint4*)&pl[wave][ml][ks * 32 + 16 + hl * 8];
#pragma unroll
      for (int j = 0; j < 4; ++j) {
        Frag16 bv;                          // B[kk=key, n=e] = vlT[e][key]
        bv.q[0] = *(const uint4*)&vlT[j * 16 + ml][ks * 32 + hl * 8];
        bv.q[1] = *(const uint4*)&vlT[j * 16 + ml][ks * 32 + 16 + hl * 8];
        oacc[j] = wmma_bf16(ap.v, bv.v, oacc[j]);
      }
    }
    __syncthreads();   // protect kl/vl/vlT before next tile's writes
  }

  // ---- Normalize and store O (f32) into o buffer at head columns.
#pragma unroll
  for (int j = 0; j < 4; ++j) {
    const int e = j * 16 + ml;
#pragma unroll
    for (int r = 0; r < 8; ++r) {
      const int row = qrow0 + hl * 8 + r;
      op[(long)row * DD + e] = oacc[j][r] / lrow[r];
    }
  }
}

// ---------------------------------------------------------------------------
// Elementwise / reduction kernels
// ---------------------------------------------------------------------------
__global__ void embed_kernel(const int* __restrict__ idx,
                             const float* __restrict__ tok,
                             const float* __restrict__ pos,
                             float* __restrict__ x)
{
  long i = (long)blockIdx.x * blockDim.x + threadIdx.x;
  if (i >= (long)BT * DD) return;
  const int d  = (int)(i & (DD - 1));
  const long bt = i >> 10;
  const int t  = (int)(bt & (TT - 1));
  const int tokid = idx[bt];
  x[i] = tok[(long)tokid * DD + d] + pos[(long)t * DD + d];
}

__global__ __launch_bounds__(256)
void ln_kernel(const float* __restrict__ x, const float* __restrict__ g,
               const float* __restrict__ beta, float* __restrict__ y)
{
  __shared__ float red[256];
  const int row = blockIdx.x;
  const float* xr = x + (long)row * DD;

  float s = 0.0f;
  for (int i = threadIdx.x; i < DD; i += 256) s += xr[i];
  red[threadIdx.x] = s; __syncthreads();
  for (int o = 128; o > 0; o >>= 1) {
    if (threadIdx.x < o) red[threadIdx.x] += red[threadIdx.x + o];
    __syncthreads();
  }
  const float mu = red[0] / (float)DD;
  __syncthreads();

  float vs = 0.0f;
  for (int i = threadIdx.x; i < DD; i += 256) { float d = xr[i] - mu; vs += d * d; }
  red[threadIdx.x] = vs; __syncthreads();
  for (int o = 128; o > 0; o >>= 1) {
    if (threadIdx.x < o) red[threadIdx.x] += red[threadIdx.x + o];
    __syncthreads();
  }
  const float rstd = rsqrtf(red[0] / (float)DD + 1e-5f);

  for (int i = threadIdx.x; i < DD; i += 256)
    y[(long)row * DD + i] = (xr[i] - mu) * rstd * g[i] + beta[i];
}

__global__ void zero_loss_kernel(float* loss) { *loss = 0.0f; }

__global__ __launch_bounds__(256)
void loss_kernel(const float* __restrict__ logits, const int* __restrict__ targets,
                 float* __restrict__ loss)
{
  __shared__ float red[256];
  const int row = blockIdx.x;                 // 0..BT-1
  const float* lr = logits + (long)row * VV;

  float mx = -1e30f;
  for (int i = threadIdx.x; i < VV; i += 256) mx = fmaxf(mx, lr[i]);
  red[threadIdx.x] = mx; __syncthreads();
  for (int o = 128; o > 0; o >>= 1) {
    if (threadIdx.x < o) red[threadIdx.x] = fmaxf(red[threadIdx.x], red[threadIdx.x + o]);
    __syncthreads();
  }
  mx = red[0];
  __syncthreads();

  float s = 0.0f;
  for (int i = threadIdx.x; i < VV; i += 256) s += __expf(lr[i] - mx);
  red[threadIdx.x] = s; __syncthreads();
  for (int o = 128; o > 0; o >>= 1) {
    if (threadIdx.x < o) red[threadIdx.x] += red[threadIdx.x + o];
    __syncthreads();
  }
  if (threadIdx.x == 0) {
    const float lse = mx + __logf(red[0]);
    atomicAdd(loss, (lse - lr[targets[row]]) / (float)BT);
  }
}

// ---------------------------------------------------------------------------
// Host-side orchestration
// ---------------------------------------------------------------------------
extern "C" void kernel_launch(void* const* d_in, const int* in_sizes, int n_in,
                              void* d_out, int out_size, void* d_ws, size_t ws_size,
                              hipStream_t stream)
{
  (void)in_sizes; (void)n_in; (void)out_size; (void)ws_size;

  const int*   idx     = (const int*)  d_in[0];
  const int*   targets = (const int*)  d_in[1];
  const float* tok_emb = (const float*)d_in[2];
  const float* pos_emb = (const float*)d_in[3];
  const float* Wq      = (const float*)d_in[4];
  const float* Wk      = (const float*)d_in[5];
  const float* Wv      = (const float*)d_in[6];
  const float* Wo      = (const float*)d_in[7];
  const float* bo      = (const float*)d_in[8];
  const float* W1      = (const float*)d_in[9];
  const float* b1      = (const float*)d_in[10];
  const float* W2      = (const float*)d_in[11];
  const float* b2      = (const float*)d_in[12];
  const float* ln_g    = (const float*)d_in[13];
  const float* ln_b    = (const float*)d_in[14];
  const float* Wf      = (const float*)d_in[15];
  const float* bf      = (const float*)d_in[16];

  const long XSZ = (long)BT * DD;          // 2M elements
  float* ws   = (float*)d_ws;
  float* x    = ws;
  float* qf   = x   + XSZ;                 // bf16 q/k/v live in float-sized slots
  float* kf   = qf  + XSZ;
  float* vf   = kf  + XSZ;
  float* o    = vf  + XSZ;
  float* xln  = o   + XSZ;
  float* hbuf = xln + XSZ;                 // BT x 4D

  // x = tok_emb[idx] + pos_emb[:T]
  embed_kernel<<<(unsigned)((XSZ + 255) / 256), 256, 0, stream>>>(idx, tok_emb, pos_emb, x);

  const long WqkvL = (long)HH * DD * HD;   // per-layer q/k/v weight size
  const float scale = 1.0f / 32.0f;        // D^-0.5 (source scales by n_embd)

  for (int l = 0; l < LL; ++l) {
    const float* wq = Wq + (long)l * WqkvL;
    const float* wk = Wk + (long)l * WqkvL;
    const float* wv = Wv + (long)l * WqkvL;
    const float* wo = Wo + (long)l * DD * DD;
    const float* bo_ = bo + (long)l * DD;
    const float* w1 = W1 + (long)l * DD * (4 * DD);
    const float* b1_ = b1 + (long)l * (4 * DD);
    const float* w2 = W2 + (long)l * (4 * DD) * DD;
    const float* b2_ = b2 + (long)l * DD;

    // Q/K/V projections: per-head batched GEMM (z = head), bf16 outputs.
    dim3 gqkv(1, BT / 64, HH);
    gemm_bf16_kernel<<<gqkv, 128, 0, stream>>>(x, wq, nullptr, nullptr, qf,
        BT, HD, DD, DD, HD, DD, 0L, (long)DD * HD, (long)HD, 0, 1);
    gemm_bf16_kernel<<<gqkv, 128, 0, stream>>>(x, wk, nullptr, nullptr, kf,
        BT, HD, DD, DD, HD, DD, 0L, (long)DD * HD, (long)HD, 0, 1);
    gemm_bf16_kernel<<<gqkv, 128, 0, stream>>>(x, wv, nullptr, nullptr, vf,
        BT, HD, DD, DD, HD, DD, 0L, (long)DD * HD, (long)HD, 0, 1);

    // Causal attention per (b,h): o = softmax(q k^T * scale) v
    attn_kernel<<<dim3(TT / 64, BB * HH), 128, 0, stream>>>(
        (const __bf16*)qf, (const __bf16*)kf, (const __bf16*)vf, o, scale);

    // x = x + o @ Wo + bo
    gemm_bf16_kernel<<<dim3(DD / 64, BT / 64, 1), 128, 0, stream>>>(o, wo, bo_, x, x,
        BT, DD, DD, DD, DD, DD, 0L, 0L, 0L, 0, 0);

    // h = relu(x @ W1 + b1)
    gemm_bf16_kernel<<<dim3((4 * DD) / 64, BT / 64, 1), 128, 0, stream>>>(x, w1, b1_, nullptr, hbuf,
        BT, 4 * DD, DD, DD, 4 * DD, 4 * DD, 0L, 0L, 0L, 1, 0);

    // x = x + h @ W2 + b2
    gemm_bf16_kernel<<<dim3(DD / 64, BT / 64, 1), 128, 0, stream>>>(hbuf, w2, b2_, x, x,
        BT, DD, 4 * DD, 4 * DD, DD, DD, 0L, 0L, 0L, 0, 0);
  }

  // Final LayerNorm
  ln_kernel<<<BT, 256, 0, stream>>>(x, ln_g, ln_b, xln);

  // logits = xln @ Wf + bf  -> written straight to d_out
  float* logits = (float*)d_out;
  gemm_bf16_kernel<<<dim3(VV / 64, BT / 64, 1), 128, 0, stream>>>(xln, Wf, bf, nullptr, logits,
      BT, VV, DD, DD, VV, VV, 0L, 0L, 0L, 0, 0);

  // loss = mean(lse(row) - logit[target])
  float* loss = logits + (long)BT * VV;
  zero_loss_kernel<<<1, 1, 0, stream>>>(loss);
  loss_kernel<<<BT, 256, 0, stream>>>(logits, targets, loss);
}
